// Moebius_Flow_20572893348082
// MI455X (gfx1250) — matrix operand order repaired and checked by reading, not compiled, not run on hardware
//
#include <hip/hip_runtime.h>
#include <hip/hip_bf16.h>
#include <math.h>

typedef __attribute__((ext_vector_type(16))) _Float16 v16h;
typedef __attribute__((ext_vector_type(8)))  float    v8f;

#define TWO_PI_F 6.28318530717958647692f

// ---------------------------------------------------------------------------
// Prep: pack W1 (64x512 f32, row-major) -> Wt1[n*64+k] f16  (N-major)
//       pack W2 (512x513 f32, row-major, drop last col) -> Wt2[n*512+k] f16
// N-major layout makes every WMMA B-operand lane-load a contiguous 32B v16h.
// ---------------------------------------------------------------------------
__global__ void moebius_pack_weights(const float* __restrict__ W1,
                                     const float* __restrict__ W2,
                                     _Float16* __restrict__ Wt1,
                                     _Float16* __restrict__ Wt2) {
    int idx = blockIdx.x * blockDim.x + threadIdx.x;
    const int NW1 = 512 * 64;
    if (idx < NW1) {
        int n = idx >> 6, k = idx & 63;
        Wt1[n * 64 + k] = (_Float16)W1[k * 512 + n];
    } else {
        idx -= NW1;
        if (idx < 512 * 512) {
            int n = idx >> 9, k = idx & 511;
            Wt2[n * 512 + k] = (_Float16)W2[k * 513 + n];
        }
    }
}

// ---------------------------------------------------------------------------
// Fused MLP (f16 WMMA) + Moebius epilogue.
// Block = 128 threads (4 waves) covering 32 batch rows (2 M-tiles of 16).
// Each wave owns 8 of the 32 N-chunks, processed in pairs -> 2x2 register
// blocking: per k-step 2 A frags + 2 B frags feed 4 WMMAs.
// N-chunk nc of GEMM2 == angle dim d (16 cols = 8 centers x 2 comps).
// ---------------------------------------------------------------------------
__launch_bounds__(128)
__global__ void moebius_fused(const float* __restrict__ theta,
                              const float* __restrict__ r,
                              const float* __restrict__ x,
                              const _Float16* __restrict__ Wt1,
                              const float* __restrict__ b1,
                              const _Float16* __restrict__ Wt2,
                              const float* __restrict__ b2,
                              const float* __restrict__ eta,
                              float* __restrict__ out, int B) {
    __shared__ __align__(32) _Float16 hT[32 * 512];   // h tile, f16 (32 KB)
    __shared__ float pbuf[4][4][16 * 17];             // per-wave 4x (16x16) D tiles

    const int tid  = threadIdx.x;
    const int wave = tid >> 5;
    const int lane = tid & 31;
    const int lh   = lane & 15;   // M (A,C/D rows) or N (B cols) index
    const int hi   = lane >> 4;   // half-wave select
    const int rowbase = blockIdx.x * 32;

    const float etav = eta[0];

    // A k order used here (contiguous, same as B): lanes 0-15 hold K 0..15, lanes 16-31 hold K 16..31
    const int kbaseA = hi * 16;
    // B k order used here (contiguous, same as A): lanes 0-15 hold K 0..15, lanes 16-31 hold K 16..31
    const int kbaseB = hi * 16;

    // ---------------- GEMM1: h = relu(x @ W1 + b1), f16 result into LDS -----
    {
        union AU { v16h v; _Float16 h[16]; };
        AU afrag[2][2];                               // [mtile][kstep], hoisted
        #pragma unroll
        for (int mt = 0; mt < 2; ++mt) {
            const float* xrow = x + (size_t)(rowbase + mt * 16 + lh) * 64;
            #pragma unroll
            for (int ks = 0; ks < 2; ++ks) {
                const float* p0 = xrow + ks * 32 + kbaseA;
                const float* p1 = p0 + 8;
                #pragma unroll
                for (int e = 0; e < 8; ++e) {
                    afrag[mt][ks].h[e]     = (_Float16)p0[e];
                    afrag[mt][ks].h[e + 8] = (_Float16)p1[e];
                }
            }
        }
        for (int nc = wave * 8; nc < wave * 8 + 8; ++nc) {
            const int n0 = nc * 16;
            v8f acc0 = {}, acc1 = {};
            #pragma unroll
            for (int ks = 0; ks < 2; ++ks) {
                v16h bm = *(const v16h*)(Wt1 + (size_t)(n0 + lh) * 64 + ks * 32 + kbaseB);
                acc0 = __builtin_amdgcn_wmma_f32_16x16x32_f16(
                           false, afrag[0][ks].v, false, bm, (short)0, acc0, false, false);
                acc1 = __builtin_amdgcn_wmma_f32_16x16x32_f16(
                           false, afrag[1][ks].v, false, bm, (short)0, acc1, false, false);
            }
            const float b1c = b1[n0 + lh];
            #pragma unroll
            for (int j = 0; j < 8; ++j) {          // D: vgpr j -> M = j + 8*hi, N = lh
                float v0 = acc0[j] + b1c;
                float v1 = acc1[j] + b1c;
                v0 = v0 > 0.f ? v0 : 0.f;
                v1 = v1 > 0.f ? v1 : 0.f;
                hT[(j + hi * 8) * 512 + n0 + lh]        = (_Float16)v0;
                hT[(16 + j + hi * 8) * 512 + n0 + lh]   = (_Float16)v1;
            }
        }
    }
    __syncthreads();   // hT columns come from all 4 waves

    // ---------------- GEMM2 (2x2 blocked) + Moebius epilogue ----------------
    for (int p = 0; p < 4; ++p) {
        const int ncA = wave * 8 + p * 2;
        const int ncB = ncA + 1;
        v8f acc00 = {}, acc01 = {}, acc10 = {}, acc11 = {};
        #pragma unroll 4
        for (int ks = 0; ks < 16; ++ks) {
            const int k0 = ks * 32;
            v16h bm0 = *(const v16h*)(Wt2 + (size_t)(ncA * 16 + lh) * 512 + k0 + kbaseB);
            v16h bm1 = *(const v16h*)(Wt2 + (size_t)(ncB * 16 + lh) * 512 + k0 + kbaseB);
            union { v16h v; uint4 q[2]; } a;       // two ds_load_b128
            a.q[0] = *(const uint4*)(&hT[lh * 512 + k0 + kbaseA]);
            a.q[1] = *(const uint4*)(&hT[lh * 512 + k0 + kbaseA + 8]);
            acc00 = __builtin_amdgcn_wmma_f32_16x16x32_f16(
                        false, a.v, false, bm0, (short)0, acc00, false, false);
            acc01 = __builtin_amdgcn_wmma_f32_16x16x32_f16(
                        false, a.v, false, bm1, (short)0, acc01, false, false);
            a.q[0] = *(const uint4*)(&hT[(16 + lh) * 512 + k0 + kbaseA]);
            a.q[1] = *(const uint4*)(&hT[(16 + lh) * 512 + k0 + kbaseA + 8]);
            acc10 = __builtin_amdgcn_wmma_f32_16x16x32_f16(
                        false, a.v, false, bm0, (short)0, acc10, false, false);
            acc11 = __builtin_amdgcn_wmma_f32_16x16x32_f16(
                        false, a.v, false, bm1, (short)0, acc11, false, false);
        }
        #pragma unroll
        for (int j = 0; j < 8; ++j) {
            const int rowi = (j + hi * 8) * 17 + lh;
            pbuf[wave][0][rowi] = acc00[j];
            pbuf[wave][1][rowi] = acc01[j];
            pbuf[wave][2][rowi] = acc10[j];
            pbuf[wave][3][rowi] = acc11[j];
        }

        // Epilogue: lane lh owns row m=lh of each tile; hi picks centers 0..3/4..7
        #pragma unroll
        for (int mt = 0; mt < 2; ++mt) {
            #pragma unroll
            for (int q = 0; q < 2; ++q) {
                const int tile = mt * 2 + q;
                const int d    = ncA + q;
                const int row  = rowbase + mt * 16 + lh;
                const float th = theta[(size_t)row * 32 + d];
                const float rr = r[(size_t)row * 32 + d];
                const float zx = rr * cosf(th);
                const float zy = rr * sinf(th);
                float sumTh = 0.f, sumSc = 0.f;
                #pragma unroll
                for (int cc = 0; cc < 4; ++cc) {
                    const int c = hi * 4 + cc;
                    float wx = (pbuf[wave][tile][lh * 17 + 2 * c]     + b2[d * 16 + 2 * c])     * etav;
                    float wy = (pbuf[wave][tile][lh * 17 + 2 * c + 1] + b2[d * 16 + 2 * c + 1]) * etav;
                    const float nw = sqrtf(wx * wx + wy * wy);
                    const float s  = rr * 0.99f / (1.f + nw);   // squash |w| < 0.99 r
                    wx *= s; wy *= s;
                    const float zwx = zx - wx, zwy = zy - wy;
                    const float dist2 = zwx * zwx + zwy * zwy;
                    const float scale = (rr * rr - (wx * wx + wy * wy)) / dist2;
                    const float zox = scale * zwx - wx;
                    const float zoy = scale * zwy - wy;
                    float tc = atan2f(zoy, zox);
                    tc = tc < 0.f ? tc + TWO_PI_F : tc;          // mod 2*pi
                    sumTh += tc;
                    sumSc += scale;
                }
                sumTh += __shfl_xor(sumTh, 16, 32);
                sumSc += __shfl_xor(sumSc, 16, 32);
                if (hi == 0) {
                    out[(size_t)row * 32 + d] = sumTh * 0.125f;
                    out[(size_t)B * 32 + (size_t)row * 32 + d] =
                        logf(sumSc * 0.125f + 1e-8f);
                }
            }
        }
    }
}

// ---------------------------------------------------------------------------
extern "C" void kernel_launch(void* const* d_in, const int* in_sizes, int n_in,
                              void* d_out, int out_size, void* d_ws, size_t ws_size,
                              hipStream_t stream) {
    const float* theta = (const float*)d_in[0];
    const float* r     = (const float*)d_in[1];
    const float* x     = (const float*)d_in[2];
    const float* W1    = (const float*)d_in[3];
    const float* b1    = (const float*)d_in[4];
    const float* W2    = (const float*)d_in[5];
    const float* b2    = (const float*)d_in[6];
    const float* eta   = (const float*)d_in[7];
    float* out = (float*)d_out;

    const int B = in_sizes[0] / 32;

    _Float16* Wt1 = (_Float16*)d_ws;                          // 512*64*2  = 64 KB
    _Float16* Wt2 = (_Float16*)((char*)d_ws + 512 * 64 * 2);  // 512*512*2 = 512 KB

    const int packN = 512 * 64 + 512 * 512;
    moebius_pack_weights<<<(packN + 255) / 256, 256, 0, stream>>>(W1, W2, Wt1, Wt2);

    moebius_fused<<<B / 32, 128, 0, stream>>>(theta, r, x, Wt1, b1, Wt2, b2,
                                              eta, out, B);
}
